// SequenceToSequence_61503931679068
// MI455X (gfx1250) — matrix operand order repaired and checked by reading, not compiled, not run on hardware
//
#include <hip/hip_runtime.h>
#include <math.h>

// Problem constants (from reference)
#define V_SZ   32000
#define E_SZ   256
#define H_SZ   512
#define S_ENC  128
#define T_DEC  64
#define B_SZ   32
#define NSTEP  (T_DEC - 1)      // 63
#define XDIM   (H_SZ + E_SZ)    // 768
#define G3H    (3 * H_SZ)       // 1536
#define NPANEL (V_SZ / 16)      // 2000 column panels of 16
#define NKCH   (H_SZ / 32)      // 16 k-chunks of 32
#define OUT_ROW_STRIDE ((size_t)NSTEP * V_SZ)   // 63*32000 per batch row of preds

typedef __attribute__((ext_vector_type(16))) __bf16 v16bf;
typedef __attribute__((ext_vector_type(8)))  float  v8f;

union Frag {
    v16bf v;
    uint4 q[2];   // 32 bytes
};

__device__ __forceinline__ unsigned short f32_to_bf16(float f) {
    unsigned int u = __float_as_uint(f);
    unsigned int r = u + 0x7FFFu + ((u >> 16) & 1u);   // round-to-nearest-even
    return (unsigned short)(r >> 16);
}

__device__ __forceinline__ float sigmoidf(float x) {
    return 1.0f / (1.0f + expf(-x));
}

// ---------------------------------------------------------------------------
// K0: pack Wo (512 x 32000 f32, row-major) into bf16 WMMA B-fragment order:
//     dst[(((p*16 + c)*32 + lane)*16) + e] = bf16(Wo[k, n])
//     with k = c*32 + (lane>=16 ? 16 : 0) + e,  n = p*16 + (lane & 15)
// ---------------------------------------------------------------------------
__global__ void pack_wo_kernel(const float* __restrict__ Wo,
                               unsigned short* __restrict__ dst) {
    int idx = blockIdx.x * 256 + threadIdx.x;          // 0 .. 512*32000-1
    int k = idx / V_SZ;
    int n = idx - k * V_SZ;
    int p = n >> 4, col = n & 15;
    int c = k >> 5, kr = k & 31;
    int lane = col + ((kr & 16) ? 16 : 0);
    int e = kr & 15;
    dst[(((size_t)(p * 16 + c) * 32 + lane) << 4) + e] = f32_to_bf16(Wo[idx]);
}

// ---------------------------------------------------------------------------
// K1: K_enc[row, a] = enc_output[row, :] @ W1[:, a],  row = b*128+s (4096 rows)
// ---------------------------------------------------------------------------
__global__ void enc_proj_kernel(const float* __restrict__ enc,
                                const float* __restrict__ W1,
                                float* __restrict__ K_enc) {
    __shared__ float eS[H_SZ];
    int row = blockIdx.x;
    int tid = threadIdx.x;
    for (int i = tid; i < H_SZ; i += 256) eS[i] = enc[(size_t)row * H_SZ + i];
    __syncthreads();
    int c0 = tid, c1 = tid + 256;
    float a0 = 0.f, a1 = 0.f;
    for (int k = 0; k < H_SZ; ++k) {
        float e = eS[k];
        const float* wr = W1 + (size_t)k * H_SZ;
        a0 += e * wr[c0];
        a1 += e * wr[c1];
    }
    K_enc[(size_t)row * H_SZ + c0] = a0;
    K_enc[(size_t)row * H_SZ + c1] = a1;
}

// ---------------------------------------------------------------------------
// K2: attention for one batch element per block (32 blocks x 256 threads)
//     s2 = h@W2 ; score_s = tanh(K_enc[b,s,:]+s2)·v_a ; softmax ; ctx = attn@enc
// ---------------------------------------------------------------------------
__global__ void attention_kernel(const float* __restrict__ h,
                                 const float* __restrict__ K_enc,
                                 const float* __restrict__ enc,
                                 const float* __restrict__ W2,
                                 const float* __restrict__ v_a,
                                 float* __restrict__ ctx) {
    __shared__ float hS[H_SZ];
    __shared__ float s2[H_SZ];
    __shared__ float vS[H_SZ];
    __shared__ float sc[S_ENC];
    __shared__ float tmp[S_ENC];
    int b = blockIdx.x;
    int tid = threadIdx.x;

    for (int i = tid; i < H_SZ; i += 256) {
        hS[i] = h[(size_t)b * H_SZ + i];
        vS[i] = v_a[i];
    }
    __syncthreads();

    for (int a = tid; a < H_SZ; a += 256) {
        float acc = 0.f;
        for (int k = 0; k < H_SZ; ++k) acc += hS[k] * W2[(size_t)k * H_SZ + a];
        s2[a] = acc;
    }
    __syncthreads();

    if (tid < S_ENC) {
        const float* kr = K_enc + ((size_t)b * S_ENC + tid) * H_SZ;
        float acc = 0.f;
        for (int a = 0; a < H_SZ; ++a) acc += tanhf(kr[a] + s2[a]) * vS[a];
        sc[tid] = acc;
        tmp[tid] = acc;
    }
    __syncthreads();
    for (int s = S_ENC / 2; s > 0; s >>= 1) {
        if (tid < s) tmp[tid] = fmaxf(tmp[tid], tmp[tid + s]);
        __syncthreads();
    }
    float mx = tmp[0];
    __syncthreads();
    if (tid < S_ENC) {
        float e = expf(sc[tid] - mx);
        sc[tid] = e;
        tmp[tid] = e;
    }
    __syncthreads();
    for (int s = S_ENC / 2; s > 0; s >>= 1) {
        if (tid < s) tmp[tid] += tmp[tid + s];
        __syncthreads();
    }
    float inv = 1.0f / tmp[0];
    __syncthreads();
    if (tid < S_ENC) sc[tid] *= inv;
    __syncthreads();

    for (int j = tid; j < H_SZ; j += 256) {
        float acc = 0.f;
        const float* er = enc + (size_t)b * S_ENC * H_SZ + j;
        for (int s = 0; s < S_ENC; ++s) acc += sc[s] * er[(size_t)s * H_SZ];
        ctx[(size_t)b * H_SZ + j] = acc;
    }
}

// ---------------------------------------------------------------------------
// K3: GRU gates z, r and the x-part of candidate (a_hh = (x@Wx + b_g)[2H:])
//     grid (32, 2): blockIdx.x = b, blockIdx.y picks j half; 1 j per thread.
// ---------------------------------------------------------------------------
__global__ void gru_zr_kernel(const float* __restrict__ ctx,
                              const float* __restrict__ embedding,
                              const int* __restrict__ dec_input,
                              const int* __restrict__ dec_target,
                              const float* __restrict__ h_in,
                              const float* __restrict__ Wx,
                              const float* __restrict__ Wh,
                              const float* __restrict__ b_g,
                              int t,
                              float* __restrict__ zb,
                              float* __restrict__ rb,
                              float* __restrict__ ahh) {
    __shared__ float xS[XDIM];
    __shared__ float hS[H_SZ];
    int b = blockIdx.x;
    int tid = threadIdx.x;
    int tok = (t == 0) ? dec_input[b] : dec_target[b * T_DEC + t];

    for (int i = tid; i < XDIM; i += 256)
        xS[i] = (i < H_SZ) ? ctx[(size_t)b * H_SZ + i]
                           : embedding[(size_t)tok * E_SZ + (i - H_SZ)];
    for (int i = tid; i < H_SZ; i += 256) hS[i] = h_in[(size_t)b * H_SZ + i];
    __syncthreads();

    int j = blockIdx.y * 256 + tid;
    float gz = b_g[j], gr = b_g[H_SZ + j], gh = b_g[2 * H_SZ + j];
    for (int k = 0; k < XDIM; ++k) {
        float xv = xS[k];
        const float* wr = Wx + (size_t)k * G3H;
        gz += xv * wr[j];
        gr += xv * wr[H_SZ + j];
        gh += xv * wr[2 * H_SZ + j];
    }
    for (int k = 0; k < H_SZ; ++k) {
        float hv = hS[k];
        const float* wr = Wh + (size_t)k * G3H;
        gz += hv * wr[j];
        gr += hv * wr[H_SZ + j];
    }
    zb[(size_t)b * H_SZ + j]  = sigmoidf(gz);
    rb[(size_t)b * H_SZ + j]  = sigmoidf(gr);
    ahh[(size_t)b * H_SZ + j] = gh;
}

// ---------------------------------------------------------------------------
// K4: candidate + state update; emits h_new (f32) and h_new (bf16 row-major
//     32x512 for the WMMA logits kernel). Optionally writes h_fin to d_out.
// ---------------------------------------------------------------------------
__global__ void gru_hh_kernel(const float* __restrict__ zb,
                              const float* __restrict__ rb,
                              const float* __restrict__ ahh,
                              const float* __restrict__ h_in,
                              const float* __restrict__ Wh,
                              float* __restrict__ h_out,
                              unsigned short* __restrict__ h_bf,
                              float* __restrict__ h_fin /* nullable */) {
    __shared__ float rh[H_SZ];
    int b = blockIdx.x;
    int tid = threadIdx.x;
    for (int i = tid; i < H_SZ; i += 256)
        rh[i] = rb[(size_t)b * H_SZ + i] * h_in[(size_t)b * H_SZ + i];
    __syncthreads();

    int j = blockIdx.y * 256 + tid;
    float acc = ahh[(size_t)b * H_SZ + j];
    for (int k = 0; k < H_SZ; ++k)
        acc += rh[k] * Wh[(size_t)k * G3H + 2 * H_SZ + j];
    float hh = tanhf(acc);
    float zv = zb[(size_t)b * H_SZ + j];
    float hv = h_in[(size_t)b * H_SZ + j];
    float hn = zv * hv + (1.0f - zv) * hh;
    h_out[(size_t)b * H_SZ + j] = hn;
    h_bf[(size_t)b * H_SZ + j]  = f32_to_bf16(hn);
    if (h_fin) h_fin[(size_t)b * H_SZ + j] = hn;
}

// ---------------------------------------------------------------------------
// K5: logits = h_new(32x512) @ Wo(512x32000) + bo via v_wmma_f32_16x16x32_bf16
//     A (h, 32 KB bf16) is async-staged to LDS once per block
//     (global_load_async_to_lds_b128 + s_wait_asynccnt) and served as
//     ds_load_b128; B streams from packed global fragments. 8 waves/block,
//     one 16-column panel per wave, 2 M-tiles, 16 K-chunks of 32.
// ---------------------------------------------------------------------------
__global__ void logits_wmma_kernel(const unsigned short* __restrict__ h_bf,
                                   const unsigned short* __restrict__ wo_frag,
                                   const float* __restrict__ bo,
                                   float* __restrict__ out,
                                   int t) {
    __shared__ unsigned short hS[B_SZ * H_SZ];   // 32 KB
    int tid = threadIdx.x;

    // Async DMA h (32x512 bf16) global -> LDS: 2048 b128 transfers, 8/thread.
    for (int i = tid; i < (B_SZ * H_SZ) / 8; i += 256) {
        unsigned lds_addr = (unsigned)(size_t)&hS[i * 8];
        const unsigned short* gp = h_bf + (size_t)i * 8;
        asm volatile("global_load_async_to_lds_b128 %0, %1, off"
                     :: "v"(lds_addr), "v"(gp)
                     : "memory");
    }
    asm volatile("s_wait_asynccnt 0" ::: "memory");
    __syncthreads();

    int lane = tid & 31;
    int p = blockIdx.x * 8 + (tid >> 5);               // panel id, 0..1999
    int col  = lane & 15;
    int koff = (lane & 16) ? 16 : 0;                   // A-fragment K offset
    int row0 = col;                                    // A rows: M = lane&15 (+16)
    int row1 = col + 16;

    v8f acc0 = {0.f, 0.f, 0.f, 0.f, 0.f, 0.f, 0.f, 0.f};
    v8f acc1 = {0.f, 0.f, 0.f, 0.f, 0.f, 0.f, 0.f, 0.f};

    const unsigned short* bbase = wo_frag + (((size_t)(p * 16) * 32 + lane) << 4);

    for (int c = 0; c < NKCH; ++c) {
        Frag a0, a1, bf;
        const unsigned short* ap0 = &hS[row0 * H_SZ + c * 32 + koff];
        const unsigned short* ap1 = &hS[row1 * H_SZ + c * 32 + koff];
        a0.q[0] = *(const uint4*)(ap0);                // ds_load_b128
        a0.q[1] = *(const uint4*)(ap0 + 8);
        a1.q[0] = *(const uint4*)(ap1);
        a1.q[1] = *(const uint4*)(ap1 + 8);
        const unsigned short* bp = bbase + (size_t)c * 512;   // 32*16 elems/chunk
        bf.q[0] = ((const uint4*)bp)[0];
        bf.q[1] = ((const uint4*)bp)[1];
        if (c + 1 < NKCH)
            __builtin_prefetch(bbase + (size_t)(c + 1) * 512, 0, 3);

        acc0 = __builtin_amdgcn_wmma_f32_16x16x32_bf16(
                   false, a0.v, false, bf.v, (short)0, acc0, false, false);
        acc1 = __builtin_amdgcn_wmma_f32_16x16x32_bf16(
                   false, a1.v, false, bf.v, (short)0, acc1, false, false);
    }

    int n = p * 16 + col;
    float bias = bo[n];
    int rbase = (lane & 16) ? 8 : 0;
    size_t tn = (size_t)t * V_SZ + n;
#pragma unroll
    for (int e = 0; e < 8; ++e) {
        int m0 = rbase + e;
        int m1 = m0 + 16;
        out[(size_t)m0 * OUT_ROW_STRIDE + tn] = acc0[e] + bias;
        out[(size_t)m1 * OUT_ROW_STRIDE + tn] = acc1[e] + bias;
    }
}

// ---------------------------------------------------------------------------
// Host-side orchestration
// ---------------------------------------------------------------------------
extern "C" void kernel_launch(void* const* d_in, const int* in_sizes, int n_in,
                              void* d_out, int out_size, void* d_ws, size_t ws_size,
                              hipStream_t stream) {
    const int*   dec_input  = (const int*)d_in[0];
    const float* dec_hidden = (const float*)d_in[1];
    const float* enc_output = (const float*)d_in[2];
    const int*   dec_target = (const int*)d_in[3];
    const float* embedding  = (const float*)d_in[4];
    const float* W1         = (const float*)d_in[5];
    const float* W2         = (const float*)d_in[6];
    const float* v_a        = (const float*)d_in[7];
    const float* Wx         = (const float*)d_in[8];
    const float* Wh         = (const float*)d_in[9];
    const float* b_g        = (const float*)d_in[10];
    const float* Wo         = (const float*)d_in[11];
    const float* bo         = (const float*)d_in[12];
    float* out = (float*)d_out;

    char* ws = (char*)d_ws;
    // Workspace layout (256B aligned offsets), ~41.6 MB total
    unsigned short* wo_frag = (unsigned short*)(ws);                 // 32,768,000 B
    float*          K_enc   = (float*)(ws + 32768000);               //  8,388,608 B
    float*          h_a     = (float*)(ws + 41156608);               //     65,536 B
    float*          h_b     = (float*)(ws + 41222144);               //     65,536 B
    unsigned short* h_bf    = (unsigned short*)(ws + 41287680);      //     32,768 B
    float*          ctx     = (float*)(ws + 41320448);               //     65,536 B
    float*          zb      = (float*)(ws + 41385984);               //     65,536 B
    float*          rb      = (float*)(ws + 41451520);               //     65,536 B
    float*          ahh     = (float*)(ws + 41517056);               //     65,536 B

    // One-time: pack Wo to bf16 fragments; hoisted enc@W1; initial attention ctx
    pack_wo_kernel<<<(H_SZ * V_SZ) / 256, 256, 0, stream>>>(Wo, wo_frag);
    enc_proj_kernel<<<B_SZ * S_ENC, 256, 0, stream>>>(enc_output, W1, K_enc);
    attention_kernel<<<B_SZ, 256, 0, stream>>>(dec_hidden, K_enc, enc_output, W2, v_a, ctx);

    float* h_fin = out + (size_t)B_SZ * NSTEP * V_SZ;   // tail of d_out

    for (int t = 0; t < NSTEP; ++t) {
        const float* h_in  = (t == 0) ? dec_hidden : ((t & 1) ? h_a : h_b);
        float*       h_out = (t & 1) ? h_b : h_a;

        gru_zr_kernel<<<dim3(B_SZ, 2), 256, 0, stream>>>(
            ctx, embedding, dec_input, dec_target, h_in, Wx, Wh, b_g, t, zb, rb, ahh);
        gru_hh_kernel<<<dim3(B_SZ, 2), 256, 0, stream>>>(
            zb, rb, ahh, h_in, Wh, h_out, h_bf,
            (t == NSTEP - 1) ? h_fin : (float*)nullptr);
        logits_wmma_kernel<<<NPANEL / 8, 256, 0, stream>>>(h_bf, wo_frag, bo, out, t);
        if (t < NSTEP - 1)
            attention_kernel<<<B_SZ, 256, 0, stream>>>(h_out, K_enc, enc_output, W2, v_a, ctx);
    }
}
